// FFQLinear_65919158059423
// MI455X (gfx1250) — matrix operand order repaired and checked by reading, not compiled, not run on hardware
//
#include <hip/hip_runtime.h>

// ---------------------------------------------------------------------------
// FFQLinear for MI455X (gfx1250):  y = x @ ((q - zp)*scale) + bias
// Rewritten as: y[m,n] = scale*(x@q)[m,n] - scale*zp*rowsum(x)[m] + bias[n]
// GEMM: M=4096 (B*S), K=4096 (DIN), N=4096 (DOUT), f16 WMMA, f32 accumulate.
// q in [0,255] is exact in f16, so x@q keeps full int-weight fidelity.
// Block tile 128x256, BK=32, 8 waves (2x4), wave tile 64x64 = 16 WMMA/K-step.
// ---------------------------------------------------------------------------

typedef __attribute__((ext_vector_type(16))) _Float16 v16h;
typedef __attribute__((ext_vector_type(8)))  float    v8f;

constexpr int MDIM = 2 * 2048;   // B*S
constexpr int NDIM = 4096;       // DOUT
constexpr int KDIM = 4096;       // DIN

constexpr int BM = 128, BN = 256, BK = 32;
constexpr int LDA = BK + 8;      // 40 halves -> 80B row stride (bank-spread)
constexpr int LDB = BK + 8;      // Bs stored [n][k] transposed

union Frag  { v16h v; uint4 q[2]; };
union Pack4 { _Float16 h[4]; uint2 u; };
union Pack2 { _Float16 h[2]; uint32_t u; };

// --- per-row sum of x (folds the zero-point term) ---------------------------
__global__ __launch_bounds__(256)
void rowsum_kernel(const float* __restrict__ x, float* __restrict__ rs) {
  __shared__ float red[256];
  const int m = blockIdx.x;
  const int t = threadIdx.x;
  const float4* xr = reinterpret_cast<const float4*>(x + (size_t)m * KDIM);
  float s = 0.f;
#pragma unroll
  for (int p = 0; p < 4; ++p) {            // 256 thr * 4 * float4 = 4096 elems
    float4 f = xr[t + p * 256];
    s += f.x + f.y + f.z + f.w;
  }
  red[t] = s;
  __syncthreads();
  for (int off = 128; off > 0; off >>= 1) {
    if (t < off) red[t] += red[t + off];
    __syncthreads();
  }
  if (t == 0) rs[m] = red[0];
}

// --- main WMMA GEMM ---------------------------------------------------------
__global__ __launch_bounds__(256)
void ffq_gemm_kernel(const float* __restrict__ x,
                     const int*   __restrict__ qw,
                     const float* __restrict__ scale,
                     const float* __restrict__ zp,
                     const float* __restrict__ bias,
                     const float* __restrict__ rowsum,
                     float*       __restrict__ out) {
  __shared__ _Float16 As[2][BM * LDA];     // [m][k] f16, 10 KB per buffer
  __shared__ _Float16 Bs[2][BN * LDB];     // [n][k] transposed, 20 KB per buffer

  const int t  = threadIdx.x;
  const int n0 = blockIdx.x * BN;
  const int m0 = blockIdx.y * BM;

  // staging coordinates
  const int arc = (t & 7) * 4;             // A col (k)  0..28
  const int arr = t >> 3;                  // A row (m)  0..31  (x4 passes)
  const int brc = (t & 63) * 4;            // B col (n)  0..252
  const int brr = (t >> 6) * 2;            // B row (k)  0,2,4,6 (x4 passes, +8)

  auto stage = [&](int buf, int k0) {
#pragma unroll
    for (int p = 0; p < 4; ++p) {          // A: 128x32 f32 -> f16
      const int r = arr + p * 32;
      const float4 f = *reinterpret_cast<const float4*>(
          x + (size_t)(m0 + r) * KDIM + k0 + arc);
      Pack4 pk;
      pk.h[0] = (_Float16)f.x; pk.h[1] = (_Float16)f.y;
      pk.h[2] = (_Float16)f.z; pk.h[3] = (_Float16)f.w;
      *reinterpret_cast<uint2*>(&As[buf][r * LDA + arc]) = pk.u;
    }
#pragma unroll
    for (int p = 0; p < 4; ++p) {          // B: 32x256 int32 -> f16, transpose
      const int r = brr + p * 8;           // even k
      const int4 q0 = *reinterpret_cast<const int4*>(
          qw + (size_t)(k0 + r) * NDIM + n0 + brc);
      const int4 q1 = *reinterpret_cast<const int4*>(
          qw + (size_t)(k0 + r + 1) * NDIM + n0 + brc);
      Pack2 pk;
#pragma unroll
      for (int i = 0; i < 4; ++i) {        // pack (k, k+1) -> aligned b32 store
        const int e0 = (i == 0) ? q0.x : (i == 1) ? q0.y : (i == 2) ? q0.z : q0.w;
        const int e1 = (i == 0) ? q1.x : (i == 1) ? q1.y : (i == 2) ? q1.z : q1.w;
        pk.h[0] = (_Float16)e0;
        pk.h[1] = (_Float16)e1;
        *reinterpret_cast<uint32_t*>(&Bs[buf][(brc + i) * LDB + r]) = pk.u;
      }
    }
  };

  // wave decomposition: 8 waves -> 2 (M) x 4 (N); wave tile 64x64 = 4x4 WMMAs
  const int lane = t & 31;
  const int wv   = t >> 5;
  const int wm   = wv & 1;
  const int wn   = wv >> 1;
  const int lhi  = lane >> 4;              // half-wave select
  const int l15  = lane & 15;
  const int kbA  = lhi * 8;                // A: K {0..7,16..23} / {8..15,24..31}
  const int kbB  = lhi * 8;                // B: K {0..7,16..23} / {8..15,24..31}

  v8f acc[4][4] = {};

  stage(0, 0);
  __syncthreads();

  constexpr int KT = KDIM / BK;            // 128 K-steps
  for (int kt = 0; kt < KT; ++kt) {
    const int cur = kt & 1;
    if (kt + 1 < KT) {
      stage(cur ^ 1, (kt + 1) * BK);
      if (kt + 2 < KT) {                   // global_prefetch_b8, two tiles ahead
        __builtin_prefetch(x  + (size_t)(m0 + arr) * KDIM + (kt + 2) * BK + arc, 0, 0);
        __builtin_prefetch(qw + (size_t)((kt + 2) * BK + brr) * NDIM + n0 + brc, 0, 0);
      }
    }

    Frag a[4], b[4];
#pragma unroll
    for (int mi = 0; mi < 4; ++mi) {       // A frags: 2x ds_load_b128 each
      const _Float16* p = &As[cur][(wm * 64 + mi * 16 + l15) * LDA + kbA];
      a[mi].q[0] = *reinterpret_cast<const uint4*>(p);
      a[mi].q[1] = *reinterpret_cast<const uint4*>(p + 16);
    }
#pragma unroll
    for (int ni = 0; ni < 4; ++ni) {       // B frags: 2x8 K halves, same order as A
      const _Float16* p = &Bs[cur][(wn * 64 + ni * 16 + l15) * LDB + kbB];
      b[ni].q[0] = *reinterpret_cast<const uint4*>(p);
      b[ni].q[1] = *reinterpret_cast<const uint4*>(p + 16);
    }

#pragma unroll
    for (int mi = 0; mi < 4; ++mi)
#pragma unroll
      for (int ni = 0; ni < 4; ++ni)
        acc[mi][ni] = __builtin_amdgcn_wmma_f32_16x16x32_f16(
            false, a[mi].v, false, b[ni].v,
            (short)0, acc[mi][ni], false, false);

    __syncthreads();
  }

  // epilogue: y = s*acc - s*zp*rowsum[m] + bias[n]  (NT stores: write-once)
  const float s   = scale[0];
  const float szp = s * zp[0];
#pragma unroll
  for (int ni = 0; ni < 4; ++ni) {
    const int n = n0 + wn * 64 + ni * 16 + l15;
    const float bn = bias[n];
#pragma unroll
    for (int mi = 0; mi < 4; ++mi) {
#pragma unroll
      for (int v = 0; v < 8; ++v) {        // C/D layout: M = v + 8*(lane>=16)
        const int m = m0 + wm * 64 + mi * 16 + lhi * 8 + v;
        const float y = s * acc[mi][ni][v] - szp * rowsum[m] + bn;
        __builtin_nontemporal_store(y, &out[(size_t)m * NDIM + n]);
      }
    }
  }
}

// ---------------------------------------------------------------------------
extern "C" void kernel_launch(void* const* d_in, const int* in_sizes, int n_in,
                              void* d_out, int out_size, void* d_ws, size_t ws_size,
                              hipStream_t stream) {
  (void)in_sizes; (void)n_in; (void)out_size; (void)ws_size;
  const float* x     = (const float*)d_in[0];
  const int*   qw    = (const int*)  d_in[1];
  const float* scale = (const float*)d_in[2];
  const float* zp    = (const float*)d_in[3];
  const float* bias  = (const float*)d_in[4];
  float* out    = (float*)d_out;
  float* rowsum = (float*)d_ws;            // 4096 floats = 16 KB scratch

  rowsum_kernel<<<MDIM, 256, 0, stream>>>(x, rowsum);
  dim3 grid(NDIM / BN, MDIM / BM);         // 16 x 32 workgroups
  ffq_gemm_kernel<<<grid, 256, 0, stream>>>(x, qw, scale, zp, bias, rowsum, out);
}
